// GAT_40638980555148
// MI455X (gfx1250) — matrix operand order repaired and checked by reading, not compiled, not run on hardware
//
#include <hip/hip_runtime.h>
#include <stdint.h>

// ---------------------------------------------------------------------------
// GAT (3x GATConv, H=8, C=64, concat=False) for MI455X / gfx1250.
//   - GEMMs: bf16 V_WMMA_F32_16X16X32_BF16, f32 accum. 16x64 strip per wave32.
//   - Softmax-aggregation: L2-resident float atomics (h fits in 192MB L2).
// ---------------------------------------------------------------------------

#define NNODES   50000
#define NEDGES   800000
#define ETOT     850000      // + NNODES self loops
#define HEADS    8
#define CH       64
#define HC       512         // HEADS*CH
#define NGRAPH   64
#define NEG_SLOPE 0.2f

typedef __attribute__((ext_vector_type(16))) __bf16    v16bf;
typedef __attribute__((ext_vector_type(8)))  float     v8f;
typedef __attribute__((ext_vector_type(4)))  unsigned  uint4v;

union AB16 { uint4v q[2]; v16bf v; };   // 32 bytes: one WMMA A/B fragment per lane

__device__ __forceinline__ uint16_t f2bf(float f) {         // f32 -> bf16 RNE
  uint32_t u = __float_as_uint(f);
  u = (u + 0x7FFFu + ((u >> 16) & 1u)) >> 16;
  return (uint16_t)u;
}
// order-preserving float<->uint for atomicMax-based segment max
__device__ __forceinline__ unsigned fenc(float f) {
  unsigned u = __float_as_uint(f);
  return (u & 0x80000000u) ? ~u : (u | 0x80000000u);
}
__device__ __forceinline__ float fdec(unsigned e) {
  unsigned u = (e & 0x80000000u) ? (e & 0x7FFFFFFFu) : ~e;
  return __uint_as_float(u);
}
__device__ __forceinline__ void edge_ends(const int* __restrict__ ei, int e, int& s, int& d) {
  if (e < NEDGES) { s = ei[e]; d = ei[NEDGES + e]; }
  else            { s = d = e - NEDGES; }                    // appended self loop
}

// ---------------- small utility kernels ----------------
__global__ void gat_zero_pool(float* __restrict__ pool) {
  int i = blockIdx.x * blockDim.x + threadIdx.x;
  if (i < NGRAPH * 3 * CH) pool[i] = 0.f;
}

__global__ void gat_conv_x(const float* __restrict__ x, uint16_t* __restrict__ xb, int n) {
  int i = blockIdx.x * blockDim.x + threadIdx.x;
  if (i < n) xb[i] = f2bf(x[i]);
}

// W[k,512] f32 -> Wt[512,K] bf16 (K-contiguous rows => 128b B-fragment loads)
__global__ void gat_pack_w(const float* __restrict__ W, uint16_t* __restrict__ Wt, int K) {
  int i = blockIdx.x * blockDim.x + threadIdx.x;
  if (i >= HC * K) return;
  int n = i / K, k = i - n * K;
  Wt[i] = f2bf(W[k * HC + n]);
}

__global__ void gat_init_layer(float* __restrict__ agg, unsigned* __restrict__ mu,
                               float* __restrict__ denom) {
  int i = blockIdx.x * blockDim.x + threadIdx.x;
  if (i < NNODES * HC) agg[i] = 0.f;
  if (i < NNODES * HEADS) { mu[i] = 0u; denom[i] = 0.f; }   // 0 < fenc(any finite)
}

// ---------------- bf16 WMMA GEMM: Hmat[N,512] = Xb[N,K] @ W[K,512] ----------------
__global__ __launch_bounds__(256) void gat_gemm_bf16(
    const uint16_t* __restrict__ Xb, const uint16_t* __restrict__ Wt,
    float* __restrict__ Hmat, int K)
{
  const int tileM   = blockIdx.x;          // 3125 tiles of 16 rows
  const int wave    = threadIdx.x >> 5;    // 8 waves -> 8 strips of 64 cols
  const int lane    = threadIdx.x & 31;
  const int mloc    = lane & 15;
  const int lhi     = lane >> 4;
  const int row     = tileM * 16 + mloc;
  const int colBase = wave * 64;

  const v8f vzero = {0.f,0.f,0.f,0.f,0.f,0.f,0.f,0.f};
  v8f acc[4] = {vzero, vzero, vzero, vzero};

  // A fragment (16x32 bf16): lanes 0-15 rows M, K 0..7 & 16..23; lanes 16-31 K 8..15 & 24..31
  const uint16_t* pa0 = Xb + row * K + 8 * lhi;
  for (int kb = 0; kb < K; kb += 32) {
    __builtin_prefetch(pa0 + kb + 64, 0, 1);               // global_prefetch_b8
    AB16 a;
    a.q[0] = *(const uint4v*)(pa0 + kb);
    a.q[1] = *(const uint4v*)(pa0 + kb + 16);
#pragma unroll
    for (int j = 0; j < 4; ++j) {
      // B fragment (32x16 bf16): lane = column, K-contiguous via pre-transposed Wt
      const uint16_t* pb = Wt + (colBase + j * 16 + mloc) * K + kb + 8 * lhi;
      AB16 b;
      b.q[0] = *(const uint4v*)pb;
      b.q[1] = *(const uint4v*)(pb + 16);
      acc[j] = __builtin_amdgcn_wmma_f32_16x16x32_bf16(
          false, a.v, false, b.v, (short)0, acc[j], false, false);
    }
  }
  // C/D layout: lanes 0-15 -> M=0..7 (VGPR r), lanes 16-31 -> M=8..15; N = lane&15
  const int rbase = (tileM * 16 + 8 * lhi) * HC + colBase + mloc;
#pragma unroll
  for (int j = 0; j < 4; ++j)
#pragma unroll
    for (int r = 0; r < 8; ++r)
      Hmat[rbase + r * HC + j * 16] = acc[j][r];
}

// ---------------- per-node attention coefficients (wave32 reductions) ----------------
__global__ __launch_bounds__(256) void gat_attn(
    const float* __restrict__ Hmat, const float* __restrict__ asrc,
    const float* __restrict__ adst, float* __restrict__ as_, float* __restrict__ ad_)
{
  const int n    = blockIdx.x;
  const int head = threadIdx.x >> 5;       // 8 heads x 32 lanes = 256
  const int lane = threadIdx.x & 31;
  const float* hp = Hmat + (n * HEADS + head) * CH;
  float h0 = hp[lane], h1 = hp[lane + 32];
  float s = h0 * asrc[head * CH + lane] + h1 * asrc[head * CH + lane + 32];
  float d = h0 * adst[head * CH + lane] + h1 * adst[head * CH + lane + 32];
#pragma unroll
  for (int off = 16; off > 0; off >>= 1) {
    s += __shfl_xor(s, off, 32);
    d += __shfl_xor(d, off, 32);
  }
  if (lane == 0) { as_[n * HEADS + head] = s; ad_[n * HEADS + head] = d; }
}

// ---------------- edge passes: segment max -> segment sum -> scatter ----------------
__global__ void gat_edge_max(const int* __restrict__ ei, const float* __restrict__ as_,
                             const float* __restrict__ ad_, unsigned* __restrict__ mu)
{
  int i = blockIdx.x * blockDim.x + threadIdx.x;
  if (i >= ETOT * HEADS) return;
  int e = i >> 3, h = i & 7, s, d;
  edge_ends(ei, e, s, d);
  float l = as_[s * HEADS + h] + ad_[d * HEADS + h];
  l = l > 0.f ? l : NEG_SLOPE * l;
  atomicMax(&mu[d * HEADS + h], fenc(l));
}

__global__ void gat_edge_denom(const int* __restrict__ ei, const float* __restrict__ as_,
                               const float* __restrict__ ad_, const unsigned* __restrict__ mu,
                               float* __restrict__ denom)
{
  int i = blockIdx.x * blockDim.x + threadIdx.x;
  if (i >= ETOT * HEADS) return;
  int e = i >> 3, h = i & 7, s, d;
  edge_ends(ei, e, s, d);
  int dh = d * HEADS + h;
  float l = as_[s * HEADS + h] + ad_[dh];
  l = l > 0.f ? l : NEG_SLOPE * l;
  atomicAdd(&denom[dh], __expf(l - fdec(mu[dh])));
}

// one wave32 per (edge, head): gather 64 floats of h[src], atomically add into agg[dst]
__global__ __launch_bounds__(256) void gat_edge_scatter(
    const int* __restrict__ ei, const float* __restrict__ as_,
    const float* __restrict__ ad_, const unsigned* __restrict__ mu,
    const float* __restrict__ denom, const float* __restrict__ Hmat,
    float* __restrict__ agg)
{
  int gid  = blockIdx.x * blockDim.x + threadIdx.x;
  int w    = gid >> 5;
  int lane = gid & 31;
  if (w >= ETOT * HEADS) return;
  int e = w >> 3, h = w & 7, s, d;
  edge_ends(ei, e, s, d);
  int dh = d * HEADS + h;
  float l = as_[s * HEADS + h] + ad_[dh];
  l = l > 0.f ? l : NEG_SLOPE * l;
  float alpha = __expf(l - fdec(mu[dh])) / denom[dh];
  const float2 hv = *(const float2*)(Hmat + (s * HEADS + h) * CH + lane * 2);
  float* q = agg + dh * CH + lane * 2;
  atomicAdd(q,     alpha * hv.x);
  atomicAdd(q + 1, alpha * hv.y);
}

// ---------------- head mean + bias + ELU + pooling (+ next-layer bf16 input) ----------------
__global__ void gat_finalize(const float* __restrict__ agg, const float* __restrict__ bias,
                             const int* __restrict__ batch, float* __restrict__ pool,
                             int layer, uint16_t* __restrict__ xb_next,
                             float* __restrict__ hout)
{
  int i = blockIdx.x * blockDim.x + threadIdx.x;
  if (i >= NNODES * CH) return;
  int n = i >> 6, c = i & 63;
  const float* p = agg + n * HC + c;
  float s = 0.f;
#pragma unroll
  for (int h = 0; h < HEADS; ++h) s += p[h * CH];
  float v = s * 0.125f + bias[c];
  v = v > 0.f ? v : (__expf(v) - 1.f);                      // ELU
  atomicAdd(&pool[batch[n] * (3 * CH) + layer * CH + c], v);
  if (xb_next) xb_next[i] = f2bf(v);
  if (hout)    hout[i] = v;
}

// ---------------------------------------------------------------------------
extern "C" void kernel_launch(void* const* d_in, const int* in_sizes, int n_in,
                              void* d_out, int out_size, void* d_ws, size_t ws_size,
                              hipStream_t stream) {
  (void)in_sizes; (void)n_in; (void)out_size; (void)ws_size;
  const float* x     = (const float*)d_in[0];
  const int*   ei    = (const int*)d_in[1];
  const int*   batch = (const int*)d_in[2];
  // d_in[3] edge_attr unused (no lin_edge in reference)
  const float* W[3]    = {(const float*)d_in[4],  (const float*)d_in[8],  (const float*)d_in[12]};
  const float* Asrc[3] = {(const float*)d_in[5],  (const float*)d_in[9],  (const float*)d_in[13]};
  const float* Adst[3] = {(const float*)d_in[6],  (const float*)d_in[10], (const float*)d_in[14]};
  const float* Bias[3] = {(const float*)d_in[7],  (const float*)d_in[11], (const float*)d_in[15]};

  char* w = (char*)d_ws;
  size_t off = 0;
  auto carve = [&](size_t bytes) { char* p = w + off; off = (off + bytes + 255) & ~(size_t)255; return p; };
  uint16_t* xbA   = (uint16_t*)carve((size_t)NNODES * 128 * 2);
  uint16_t* xbB   = (uint16_t*)carve((size_t)NNODES * CH * 2);
  uint16_t* Wt    = (uint16_t*)carve((size_t)HC * 128 * 2);
  float*    hmat  = (float*)   carve((size_t)NNODES * HC * 4);
  float*    agg   = (float*)   carve((size_t)NNODES * HC * 4);
  float*    a_s   = (float*)   carve((size_t)NNODES * HEADS * 4);
  float*    a_d   = (float*)   carve((size_t)NNODES * HEADS * 4);
  unsigned* m_u   = (unsigned*)carve((size_t)NNODES * HEADS * 4);
  float*    denom = (float*)   carve((size_t)NNODES * HEADS * 4);

  float* out = (float*)d_out;                 // [64*192 pooled | 50000*64 final h]

  gat_zero_pool<<<(NGRAPH * 3 * CH + 255) / 256, 256, 0, stream>>>(out);
  gat_conv_x<<<(NNODES * 128 + 255) / 256, 256, 0, stream>>>(x, xbA, NNODES * 128);

  for (int layer = 0; layer < 3; ++layer) {
    const int K = (layer == 0) ? 128 : CH;
    const uint16_t* xin    = (layer == 1) ? xbB : xbA;
    uint16_t*       xnext  = (layer == 0) ? xbB : (layer == 1 ? xbA : nullptr);
    float*          hfinal = (layer == 2) ? (out + NGRAPH * 3 * CH) : nullptr;

    gat_pack_w<<<(HC * K + 255) / 256, 256, 0, stream>>>(W[layer], Wt, K);
    gat_gemm_bf16<<<NNODES / 16, 256, 0, stream>>>(xin, Wt, hmat, K);
    gat_attn<<<NNODES, 256, 0, stream>>>(hmat, Asrc[layer], Adst[layer], a_s, a_d);
    gat_init_layer<<<(NNODES * HC + 255) / 256, 256, 0, stream>>>(agg, m_u, denom);
    gat_edge_max<<<(ETOT * HEADS + 255) / 256, 256, 0, stream>>>(ei, a_s, a_d, m_u);
    gat_edge_denom<<<(ETOT * HEADS + 255) / 256, 256, 0, stream>>>(ei, a_s, a_d, m_u, denom);
    gat_edge_scatter<<<ETOT, 256, 0, stream>>>(ei, a_s, a_d, m_u, denom, hmat, agg);
    gat_finalize<<<(NNODES * CH + 255) / 256, 256, 0, stream>>>(
        agg, Bias[layer], batch, out, layer, xnext, hfinal);
  }
}
